// DotProductAttention_14319420964896
// MI455X (gfx1250) — matrix operand order repaired and checked by reading, not compiled, not run on hardware
//
#include <hip/hip_runtime.h>
#include <hip/hip_bf16.h>
#include <math.h>

// Problem constants (reference: BH=32, N=2048, D=128, ALPHA=0.99, RATIO=0.75)
#define BH_    32
#define NN     2048
#define DD     128
#define ROWS   16          // query rows per workgroup (one 16-row WMMA M-tile)
#define WAVES  8           // 256 threads, wave32
#define SSTR   (NN + 4)    // padded LDS row stride (dwords) -> conflict-free PV reads
#define KSEL   1536        // int(N * 0.75)
#define ALPHA_F 0.99f
#define EPS_F   1e-10f

typedef __attribute__((ext_vector_type(16))) __bf16 v16bf;
typedef __attribute__((ext_vector_type(8)))  float  v8f;

// order-preserving float<->uint key (unsigned ascending == float ascending)
__device__ __forceinline__ unsigned ford(float f) {
    unsigned u = __float_as_uint(f);
    unsigned m = (u & 0x80000000u) ? 0xFFFFFFFFu : 0x80000000u;
    return u ^ m;
}
__device__ __forceinline__ float funord(unsigned k) {
    unsigned u = (k & 0x80000000u) ? (k & 0x7FFFFFFFu) : ~k;
    return __uint_as_float(u);
}

__device__ __forceinline__ float wredmax(float x) {
    #pragma unroll
    for (int o = 16; o > 0; o >>= 1) x = fmaxf(x, __shfl_xor(x, o, 32));
    return x;
}
__device__ __forceinline__ float wredsum(float x) {
    #pragma unroll
    for (int o = 16; o > 0; o >>= 1) x += __shfl_xor(x, o, 32);
    return x;
}

extern "C" __global__ __launch_bounds__(256, 1)
void topk_attn_gfx1250(const float* __restrict__ q,
                       const float* __restrict__ k,
                       const float* __restrict__ v,
                       float* __restrict__ out)
{
    extern __shared__ char smem[];
    float*    S    = reinterpret_cast<float*>(smem);          // ROWS x SSTR fp32
    float*    bias = S + ROWS * SSTR;                          // NN fp32
    unsigned* hist = reinterpret_cast<unsigned*>(bias + NN);   // WAVES x 256
    float*    invs = reinterpret_cast<float*>(hist + WAVES * 256); // ROWS

    const int b    = blockIdx.y;
    const int n0   = blockIdx.x * ROWS;
    const int tid  = threadIdx.x;
    const int w    = tid >> 5;
    const int lane = tid & 31;
    const int lhalf= (lane >= 16);
    const int l15  = lane & 15;

    const float* qb = q + (size_t)b * NN * DD;
    const float* kb = k + (size_t)b * NN * DD;
    const float* vb = v + (size_t)b * NN * DD;

    // ---------------- bias[m] = log(alpha^(N-1-m) + eps) ----------------
    {
        const float lga = __logf(ALPHA_F);
        for (int m = tid; m < NN; m += 256) {
            float dec = __expf((float)(NN - 1 - m) * lga);
            bias[m] = __logf(dec + EPS_F);
        }
    }
    __syncthreads();

    // ---------------- Phase A: S = Q K^T + bias  (bf16 WMMA, f32 accum) --
    // A-fragments: 16x32 bf16 per 32-dim chunk of D (4 chunks), shared by all
    // column tiles of this wave. Contiguous k order: lane half h holds k = 16h..16h+15.
    v16bf aF[4];
    {
        const float* qrow = qb + (size_t)(n0 + l15) * DD;
        #pragma unroll
        for (int kc = 0; kc < 4; ++kc) {
            int base = kc * 32 + (lhalf ? 16 : 0);
            const float4* p0 = reinterpret_cast<const float4*>(qrow + base);
            float4 a0 = p0[0], a1 = p0[1];
            const float4* p1 = reinterpret_cast<const float4*>(qrow + base + 8);
            float4 a2 = p1[0], a3 = p1[1];
            v16bf af;
            af[0]=(__bf16)a0.x; af[1]=(__bf16)a0.y; af[2]=(__bf16)a0.z; af[3]=(__bf16)a0.w;
            af[4]=(__bf16)a1.x; af[5]=(__bf16)a1.y; af[6]=(__bf16)a1.z; af[7]=(__bf16)a1.w;
            af[8]=(__bf16)a2.x; af[9]=(__bf16)a2.y; af[10]=(__bf16)a2.z; af[11]=(__bf16)a2.w;
            af[12]=(__bf16)a3.x; af[13]=(__bf16)a3.y; af[14]=(__bf16)a3.z; af[15]=(__bf16)a3.w;
            aF[kc] = af;
        }
    }

    for (int cb = 0; cb < 16; ++cb) {           // 16 column tiles of 16 per wave
        const int m0 = w * 256 + cb * 16;
        v8f acc = {};
        #pragma unroll
        for (int kc = 0; kc < 4; ++kc) {
            // B 32x16 bf16: lane n (0..15) holds K=0..15, lanes 16..31 K=16..31
            // => 16 contiguous fp32 of one k-row per lane (row-major friendly).
            const int brow  = m0 + l15;
            const int dbase = kc * 32 + (lhalf ? 16 : 0);
            const float4* kp = reinterpret_cast<const float4*>(kb + (size_t)brow * DD + dbase);
            float4 b0 = kp[0], b1 = kp[1], b2 = kp[2], b3 = kp[3];
            v16bf bfv;
            bfv[0]=(__bf16)b0.x; bfv[1]=(__bf16)b0.y; bfv[2]=(__bf16)b0.z; bfv[3]=(__bf16)b0.w;
            bfv[4]=(__bf16)b1.x; bfv[5]=(__bf16)b1.y; bfv[6]=(__bf16)b1.z; bfv[7]=(__bf16)b1.w;
            bfv[8]=(__bf16)b2.x; bfv[9]=(__bf16)b2.y; bfv[10]=(__bf16)b2.z; bfv[11]=(__bf16)b2.w;
            bfv[12]=(__bf16)b3.x; bfv[13]=(__bf16)b3.y; bfv[14]=(__bf16)b3.z; bfv[15]=(__bf16)b3.w;
            acc = __builtin_amdgcn_wmma_f32_16x16x32_bf16(
                      false, aF[kc], false, bfv, (short)0, acc, false, false);
        }
        // C layout: VGPR j -> (M=j, N=lane) lanes 0-15 ; (M=8+j, N=lane-16)
        const int col   = m0 + l15;
        const int rbase = lhalf ? 8 : 0;
        const float bc  = bias[col];
        #pragma unroll
        for (int j = 0; j < 8; ++j)
            S[(rbase + j) * SSTR + col] = acc[j] + bc;
    }
    __syncthreads();

    // ------- Phase B+C: exact 1536-th-largest via radix select + softmax ----
    // Each wave owns 2 rows; loop structure is uniform across waves so
    // __syncthreads() is legal (used only to order wave-private LDS traffic).
    for (int rr = 0; rr < 2; ++rr) {
        const int row = w * 2 + rr;
        float* srow = S + (size_t)row * SSTR;
        unsigned* h = hist + w * 256;

        unsigned prefix = 0;
        int target = KSEL;                      // rank counted from the top
        #pragma unroll
        for (int pass = 3; pass >= 0; --pass) {
            const int shift = pass * 8;
            #pragma unroll
            for (int t = 0; t < 8; ++t) h[lane * 8 + t] = 0u;
            __syncthreads();
            for (int j = 0; j < NN / 32; ++j) {
                unsigned key = ford(srow[j * 32 + lane]);
                bool match = (pass == 3) || (((key ^ prefix) >> (shift + 8)) == 0u);
                if (match) atomicAdd(&h[(key >> shift) & 255u], 1u);
            }
            __syncthreads();
            // descending scan: lane i owns bins [255-8i .. 248-8i]
            const int btop = 255 - 8 * lane;
            unsigned part = 0;
            #pragma unroll
            for (int t = 0; t < 8; ++t) part += h[btop - t];
            unsigned incl = part;
            #pragma unroll
            for (int off = 1; off < 32; off <<= 1) {
                unsigned tv = __shfl_up(incl, off, 32);
                if (lane >= off) incl += tv;
            }
            const unsigned excl = incl - part;
            const bool cond = (excl < (unsigned)target) && ((unsigned)target <= incl);
            unsigned long long bal = __ballot(cond);
            const int winner = __ffsll(bal) - 1;
            unsigned nprefix = prefix; int ntarget = target;
            if (lane == winner) {
                unsigned run = excl;
                #pragma unroll
                for (int t = 0; t < 8; ++t) {
                    unsigned c = h[btop - t];
                    if (run + c >= (unsigned)target) {
                        nprefix = prefix | ((unsigned)(btop - t) << shift);
                        ntarget = target - (int)run;
                        break;
                    }
                    run += c;
                }
            }
            prefix = (unsigned)__shfl((int)nprefix, winner, 32);
            target = __shfl(ntarget, winner, 32);
            __syncthreads();
        }
        const float T = funord(prefix);         // exact k-th largest value

        // masked softmax in place (fp32); 1/sum deferred to the output store
        float mx = -INFINITY;
        for (int j = 0; j < NN / 32; ++j)
            mx = fmaxf(mx, srow[j * 32 + lane]);
        mx = wredmax(mx);
        float sum = 0.f;
        for (int j = 0; j < NN / 32; ++j) {
            const int m = j * 32 + lane;
            float s = srow[m];
            float p = (s >= T) ? __expf(s - mx) : 0.f;
            srow[m] = p;
            sum += p;
        }
        sum = wredsum(sum);
        if (lane == 0) invs[row] = 1.f / sum;
    }
    __syncthreads();

    // ---------------- Phase D: O = P V  (bf16 WMMA, f32 accum) ----------
    {
        const int d0 = w * 16;                  // each wave owns 16 of 128 dims
        v8f acc = {};
        const float* srow = S + (size_t)l15 * SSTR;
        for (int mc = 0; mc < NN; mc += 32) {
            // A 16x32 from prob tile in LDS (convert fp32 -> bf16)
            const int moff = mc + (lhalf ? 16 : 0);
            v16bf aP;
            #pragma unroll
            for (int t = 0; t < 8; ++t) aP[t]     = (__bf16)srow[moff + t];
            #pragma unroll
            for (int t = 0; t < 8; ++t) aP[8 + t] = (__bf16)srow[moff + 8 + t];
            // B 32x16 from V: lane n holds column d0+n, K = 16 m-rows
            const int mb = mc + (lhalf ? 16 : 0);
            const float* vp = vb + (size_t)mb * DD + (d0 + l15);
            v16bf bP;
            #pragma unroll
            for (int t = 0; t < 16; ++t) bP[t] = (__bf16)vp[t * DD];
            acc = __builtin_amdgcn_wmma_f32_16x16x32_bf16(
                      false, aP, false, bP, (short)0, acc, false, false);
        }
        const int rbase = lhalf ? 8 : 0;
        const int col   = d0 + l15;
        float* ob = out + ((size_t)b * NN + n0) * DD;
        #pragma unroll
        for (int j = 0; j < 8; ++j) {
            const int r = rbase + j;
            ob[(size_t)r * DD + col] = acc[j] * invs[r];
        }
    }
}

extern "C" void kernel_launch(void* const* d_in, const int* in_sizes, int n_in,
                              void* d_out, int out_size, void* d_ws, size_t ws_size,
                              hipStream_t stream) {
    (void)in_sizes; (void)n_in; (void)out_size; (void)d_ws; (void)ws_size;
    const float* q = (const float*)d_in[0];
    const float* k = (const float*)d_in[1];
    const float* v = (const float*)d_in[2];
    float* out = (float*)d_out;

    const size_t smem = (size_t)(ROWS * SSTR) * 4     // score/prob tile
                      + (size_t)NN * 4                // bias table
                      + (size_t)WAVES * 256 * 4       // radix histograms
                      + (size_t)ROWS * 4;             // row 1/sum
    // CDNA5: up to 320KB LDS per workgroup; raise the dynamic-LDS cap.
    (void)hipFuncSetAttribute((const void*)topk_attn_gfx1250,
                              hipFuncAttributeMaxDynamicSharedMemorySize,
                              (int)smem);
    dim3 grid(NN / ROWS, BH_);
    topk_attn_gfx1250<<<grid, 256, smem, stream>>>(q, k, v, out);
}
